// Model_78194174591405
// MI455X (gfx1250) — matrix operand order repaired and checked by reading, not compiled, not run on hardware
//
#include <hip/hip_runtime.h>
#include <math.h>

// ---------------- model constants (mirror reference) ----------------
#define NNODE 350
#define SEQL  48
#define LLBL  24
#define LPRD  24
#define LDEC  72     // LLBL label steps + 48 zero-init steps
#define DM    128
#define NH    8
#define HD    16
#define DFF   512
#define CINC  5
#define MRK   4
#define MAVG  25

typedef __attribute__((ext_vector_type(16))) _Float16 v16h;
typedef __attribute__((ext_vector_type(4)))  _Float16 v4h;
typedef __attribute__((ext_vector_type(8)))  float    v8f;

// =====================================================================
// WMMA tiled GEMM:  Y[M,N] = epilogue( A[M,K] * B + bias )
//   - block = 4 waves in 2x2; block tile 32(M) x 64(N); 2 WMMA / wave / K-step
//   - A fp32 row-major (lda); B via compile-time (TRB, BS) accessor
//   - staged to LDS as f16 with branchless clamped float4 loads
//   - compile-time epilogue: HB (+bias), ACT (0 none / 2 gelu), HR (+res),
//     HA (+=Y). Requirements (all call sites satisfy): K%32==0, N%64==0,
//     lda/ldb%4==0; only M may be partial (guarded slow path).
// =====================================================================
template<int ACT, int TRB, int BS, int HB, int HR, int HA>
__global__ __launch_bounds__(128)
void k_gemm(const float* __restrict__ Ap, const float* __restrict__ Bp,
            const float* __restrict__ bias, const float* __restrict__ res,
            float* __restrict__ Yp,
            int M, int N, int K, int lda, int ldb, int ldy,
            long long aB, long long yB, long long rB)
{
  const int bz = blockIdx.z;
  const float* A = Ap + (long long)bz * aB;
  float*       Y = Yp + (long long)bz * yB;
  const float* R = HR ? (res + (long long)bz * rB) : (const float*)0;

  const int m0 = blockIdx.y * 32;
  const int n0 = blockIdx.x * 64;
  const int tid  = threadIdx.x;
  const int wave = tid >> 5, lane = tid & 31;
  const int wr = wave >> 1, wc = wave & 1;       // 2x2 wave grid
  const int g = lane >> 4, mn = lane & 15;

  __shared__ _Float16 sA[32 * 32];   // [m][k]
  __shared__ _Float16 sB[64 * 32];   // [n][k] (transposed tile)

  v8f acc0 = {}, acc1 = {};

  const int rA  = tid >> 3;          // 0..15
  const int c4  = (tid & 7) << 2;    // 0,4,..,28

  for (int k0 = 0; k0 < K; k0 += 32) {
    // ---- stage A (32x32), branchless row clamp + zero select ----
#pragma unroll
    for (int it = 0; it < 2; ++it) {
      int r   = rA + it * 16;
      int row = m0 + r;
      int rr  = row < M ? row : (M - 1);
      const float* ap = A + (long long)rr * lda + (k0 + c4);
      float4 v = *(const float4*)ap;
      float mz = (row < M) ? 1.f : 0.f;
      v4h h;
      h[0] = (_Float16)(v.x * mz); h[1] = (_Float16)(v.y * mz);
      h[2] = (_Float16)(v.z * mz); h[3] = (_Float16)(v.w * mz);
      *(v4h*)&sA[r * 32 + c4] = h;
    }
    // ---- stage B (64x32 as [n][k]) ----
    if (TRB && BS == 1) {            // B is [N,K] row-major: contiguous along k
#pragma unroll
      for (int it = 0; it < 4; ++it) {
        int r = rA + it * 16;        // n within tile, 0..63
        const float* bp = Bp + (long long)(n0 + r) * ldb + (k0 + c4);
        float4 v = *(const float4*)bp;
        v4h h;
        h[0] = (_Float16)v.x; h[1] = (_Float16)v.y;
        h[2] = (_Float16)v.z; h[3] = (_Float16)v.w;
        *(v4h*)&sB[r * 32 + c4] = h;
      }
    } else if (TRB) {                // B is [N,K] with element stride 3 (conv tap)
#pragma unroll 2
      for (int i = tid; i < 64 * 32; i += 128) {
        int n = i >> 5, cc = i & 31;
        sB[i] = (_Float16)Bp[((long long)(n0 + n) * ldb + (k0 + cc)) * 3];
      }
    } else {                         // B is [K,N] row-major: contiguous along n
#pragma unroll
      for (int it = 0; it < 4; ++it) {
        int idx = tid + it * 128;    // 0..511 float4 groups
        int kk  = idx >> 4;          // 0..31
        int cn  = (idx & 15) << 2;   // 0..60
        const float* bp = Bp + (long long)(k0 + kk) * ldb + (n0 + cn);
        float4 v = *(const float4*)bp;
        sB[(cn + 0) * 32 + kk] = (_Float16)v.x;
        sB[(cn + 1) * 32 + kk] = (_Float16)v.y;
        sB[(cn + 2) * 32 + kk] = (_Float16)v.z;
        sB[(cn + 3) * 32 + kk] = (_Float16)v.w;
      }
    }
    __syncthreads();

    // ---- fragments: contiguous k = 16*g + h in A and B; contiguous LDS (ds_load_b128) ----
    v16h a, b0, b1;
#pragma unroll
    for (int h = 0; h < 16; ++h) {
      int ka = g * 16 + h;
      a[h]  = sA[(wr * 16 + mn) * 32 + ka];
      int kb = g * 16 + h;
      b0[h] = sB[(wc * 32 + mn) * 32 + kb];
      b1[h] = sB[(wc * 32 + 16 + mn) * 32 + kb];
    }
    acc0 = __builtin_amdgcn_wmma_f32_16x16x32_f16(false, a, false, b0, (short)0, acc0, false, false);
    acc1 = __builtin_amdgcn_wmma_f32_16x16x32_f16(false, a, false, b1, (short)0, acc1, false, false);
    __syncthreads();
  }

  // ---- epilogue (compile-time configured; full-tile fast path) ----
  const int row0  = m0 + wr * 16 + g * 8;
  const bool full = (m0 + 32 <= M);
#pragma unroll
  for (int t = 0; t < 2; ++t) {
    const int col = n0 + wc * 32 + t * 16 + mn;      // always < N (N%64==0)
    v8f av = t ? acc1 : acc0;
    float bv = HB ? bias[col] : 0.f;
    long long base = (long long)row0 * ldy + col;
    if (full) {
      float rv[8], yv[8];
      if (HR) {
#pragma unroll
        for (int r = 0; r < 8; ++r) rv[r] = R[base + (long long)r * ldy];
      }
      if (HA) {
#pragma unroll
        for (int r = 0; r < 8; ++r) yv[r] = Y[base + (long long)r * ldy];
      }
#pragma unroll
      for (int r = 0; r < 8; ++r) {
        float v = av[r];
        if (HB) v += bv;
        if (ACT == 2) v = 0.5f * v * (1.f + erff(v * 0.70710678118f));
        if (HR) v += rv[r];
        if (HA) v += yv[r];
        Y[base + (long long)r * ldy] = v;
      }
    } else {
#pragma unroll
      for (int r = 0; r < 8; ++r) {
        if (row0 + r < M) {
          float v = av[r];
          if (HB) v += bv;
          if (ACT == 2) v = 0.5f * v * (1.f + erff(v * 0.70710678118f));
          long long idx = base + (long long)r * ldy;
          if (HR) v += R[idx];
          if (HA) v += Y[idx];
          Y[idx] = v;
        }
      }
    }
  }
}

// =====================================================================
// elementwise / small kernels
// =====================================================================
__global__ void k_fill(float* p, long long n, float v) {
  long long i = (long long)blockIdx.x * 256 + threadIdx.x;
  if (i < n) p[i] = v;
}
__global__ void k_add(const float* a, const float* b, float* o, long long n) {
  long long i = (long long)blockIdx.x * 256 + threadIdx.x;
  if (i < n) o[i] = a[i] + b[i];
}
__global__ void k_addrelu(const float* a, const float* b, float* o, long long n) {
  long long i = (long long)blockIdx.x * 256 + threadIdx.x;
  if (i < n) o[i] = fmaxf(a[i] + b[i], 0.f);
}
__global__ void k_relu(float* p, long long n) {
  long long i = (long long)blockIdx.x * 256 + threadIdx.x;
  if (i < n) p[i] = fmaxf(p[i], 0.f);
}

// RevIN normalize per column (col = node*5 + c) over SEQL
__global__ void k_prep(const float* __restrict__ xenc, const float* aw, const float* ab,
                       float* xeN, float* cmean, float* cstd, float* cmn2)
{
  int col = blockIdx.x * 256 + threadIdx.x;
  if (col >= NNODE * CINC) return;
  int n = col / CINC, c = col % CINC;
  const float* xp = xenc + (long long)n * SEQL * CINC + c;
  float m = 0.f;
  for (int t = 0; t < SEQL; ++t) m += xp[t * CINC];
  m /= SEQL;
  float v = 0.f;
  for (int t = 0; t < SEQL; ++t) { float d = xp[t * CINC] - m; v += d * d; }
  v /= SEQL;
  float sd = sqrtf(v + 1e-5f);
  float w = aw[c], b = ab[c];
  float m2 = 0.f;
  for (int t = 0; t < SEQL; ++t) {
    float z = (xp[t * CINC] - m) / sd * w + b;
    xeN[((long long)n * SEQL + t) * CINC + c] = z;
    m2 += z;
  }
  cmean[col] = m; cstd[col] = sd; cmn2[col] = m2 / SEQL;
}

// series_decomp: moving mean (win=25, replicate pad); per (row,d) column
__global__ void k_decomp(const float* __restrict__ x, float* seas, float* trend,
                         float* tacc, int R, int L, int D)
{
  long long i = (long long)blockIdx.x * 256 + threadIdx.x;
  if (i >= (long long)R * D) return;
  int r = (int)(i / D), d = (int)(i % D);
  const float* xp = x + (long long)r * L * D + d;
  const int p = (MAVG - 1) / 2;
  for (int t = 0; t < L; ++t) {
    float s = 0.f;
    for (int j = -p; j <= p; ++j) {
      int tt = t + j; tt = tt < 0 ? 0 : (tt >= L ? L - 1 : tt);
      s += xp[(long long)tt * D];
    }
    float tr = s / (float)MAVG;
    long long idx = (long long)r * L * D + (long long)t * D + d;
    seas[idx] = x[idx] - tr;
    if (trend) trend[idx] = tr;
    if (tacc)  tacc[idx] += tr;
  }
}

// build seasonal_init + trend_acc init  [350,72,5]
__global__ void k_seasinit(const float* seasN, const float* trndN, const float* cmn2,
                           float* sinit, float* tacc5)
{
  long long i = (long long)blockIdx.x * 256 + threadIdx.x;
  if (i >= (long long)NNODE * LDEC * CINC) return;
  int c = (int)(i % CINC); long long r = i / CINC;
  int t = (int)(r % LDEC); int n = (int)(r / LDEC);
  float s, tr;
  if (t < LLBL) {
    long long src = ((long long)n * SEQL + (SEQL - LLBL + t)) * CINC + c;
    s = seasN[src]; tr = trndN[src];
  } else { s = 0.f; tr = cmn2[n * CINC + c]; }
  sinit[i] = s; tacc5[i] = tr;
}

__global__ void k_markd(const float* me, const float* md, float* out)
{
  int i = blockIdx.x * 256 + threadIdx.x;
  if (i >= LDEC * MRK) return;
  int t = i / MRK, m = i % MRK;
  out[i] = (t < LLBL) ? me[(SEQL - LLBL + t) * MRK + m] : md[(t - LLBL) * MRK + m];
}

// embedding: circular token conv (K=5,k=3) + mark proj + PE + node emb
__global__ void k_embed(const float* __restrict__ x, const float* __restrict__ Wtok,
                        const float* __restrict__ mark, const float* __restrict__ Wmark,
                        const float* __restrict__ nodeE, float* out, int L)
{
  long long i = (long long)blockIdx.x * 256 + threadIdx.x;
  if (i >= (long long)NNODE * L * DM) return;
  int d = (int)(i % DM); long long r = i / DM;
  int t = (int)(r % L); int n = (int)(r / L);
  float s = 0.f;
  for (int j = 0; j < 3; ++j) {
    int tt = (t - 1 + j + L) % L;
    const float* xp = x + ((long long)n * L + tt) * CINC;
    for (int c = 0; c < CINC; ++c) s += xp[c] * Wtok[(d * CINC + c) * 3 + j];
  }
  for (int m = 0; m < MRK; ++m) s += mark[t * MRK + m] * Wmark[m * DM + d];
  float dv = expf(-logf(10000.f) * (float)(d & ~1) / (float)DM);
  s += (d & 1) ? cosf((float)t * dv) : sinf((float)t * dv);
  s += nodeE[n * DM + d];
  out[i] = s;
}

// [B*Lsrc,DM] token -> [B,DM,Lpad] channel-major (zero pad tail)
__global__ void k_split_t(const float* __restrict__ tok, float* dst, int B, int Lsrc, int Lpad)
{
  long long i = (long long)blockIdx.x * 256 + threadIdx.x;
  if (i >= (long long)B * DM * Lpad) return;
  int l = (int)(i % Lpad); long long r = i / Lpad;
  int d = (int)(r % DM); int b = (int)(r / DM);
  dst[i] = (l < Lsrc) ? tok[((long long)b * Lsrc + l) * DM + d] : 0.f;
}

// token -> [G, DM*L, Nn] spatial channel-major; node = g*gs + n*ns
__global__ void k_split_s(const float* __restrict__ tok, float* dst,
                          int Gd, int Nn, int L, int gs, int ns)
{
  long long i = (long long)blockIdx.x * 256 + threadIdx.x;
  if (i >= (long long)Gd * DM * L * Nn) return;
  int n = (int)(i % Nn); long long r = i / Nn;
  int l = (int)(r % L); r /= L;
  int d = (int)(r % DM); int g = (int)(r / DM);
  int node = g * gs + n * ns;
  dst[i] = tok[((long long)node * L + l) * DM + d];
}

// token -> [G*NH, Nn, HD*L] (TCN position-major layout)
__global__ void k_split_vT(const float* __restrict__ tok, float* dst,
                           int Gd, int Nn, int L, int gs, int ns)
{
  long long i = (long long)blockIdx.x * 256 + threadIdx.x;
  if (i >= (long long)Gd * NH * Nn * HD * L) return;
  int l = (int)(i % L); long long r = i / L;
  int c = (int)(r % HD); r /= HD;
  int n = (int)(r % Nn); r /= Nn;
  int h = (int)(r % NH); int g = (int)(r / NH);
  int node = g * gs + n * ns;
  dst[i] = tok[((long long)node * L + l) * DM + (h * HD + c)];
}

// circular lag-correlation, mean over channels: mc[b,tau]
__global__ void k_corr(const float* __restrict__ q, const float* __restrict__ k,
                       float* mc, int CH, int L)
{
  int b = blockIdx.x;
  __shared__ float sc[128];
  for (int i = threadIdx.x; i < L; i += blockDim.x) sc[i] = 0.f;
  __syncthreads();
  for (int ch = threadIdx.x; ch < CH; ch += blockDim.x) {
    const float* qp = q + ((long long)b * CH + ch) * L;
    const float* kp = k + ((long long)b * CH + ch) * L;
    for (int tau = 0; tau < L; ++tau) {
      float s = 0.f;
      for (int t = 0; t < L; ++t) { int tt = t + tau; if (tt >= L) tt -= L; s += qp[t] * kp[tt]; }
      atomicAdd(&sc[tau], s);
    }
  }
  __syncthreads();
  for (int i = threadIdx.x; i < L; i += blockDim.x) mc[(long long)b * L + i] = sc[i] / (float)CH;
}

__global__ void k_topk(const float* __restrict__ mc, int L, int Kt, int* dly, float* wts)
{
  int b = blockIdx.x;
  if (threadIdx.x != 0) return;
  const float* m = mc + (long long)b * L;
  int chosen[4]; float val[4];
  for (int kk = 0; kk < Kt; ++kk) {
    float best = -3e38f; int bi = 0;
    for (int i = 0; i < L; ++i) {
      bool used = false;
      for (int j = 0; j < kk; ++j) if (chosen[j] == i) used = true;
      if (!used && m[i] > best) { best = m[i]; bi = i; }
    }
    chosen[kk] = bi; val[kk] = best;
  }
  float mx = val[0];
  for (int kk = 1; kk < Kt; ++kk) mx = fmaxf(mx, val[kk]);
  float se = 0.f;
  for (int kk = 0; kk < Kt; ++kk) { val[kk] = expf(val[kk] - mx); se += val[kk]; }
  for (int kk = 0; kk < Kt; ++kk) { dly[b * 4 + kk] = chosen[kk]; wts[b * 4 + kk] = val[kk] / se; }
}

// temporal delay aggregation -> token layout
__global__ void k_gather_t(const float* __restrict__ vs, const int* dly, const float* wts,
                           float* out, int B, int L, int Kt)
{
  long long i = (long long)blockIdx.x * 256 + threadIdx.x;
  if (i >= (long long)B * L * DM) return;
  int d = (int)(i % DM); long long r = i / DM;
  int l = (int)(r % L); int b = (int)(r / L);
  const float* vp = vs + ((long long)b * DM + d) * L;
  float s = 0.f;
  for (int kk = 0; kk < Kt; ++kk) {
    int ll = l + dly[b * 4 + kk]; if (ll >= L) ll -= L;
    s += wts[b * 4 + kk] * vp[ll];
  }
  out[i] = s;
}

// spatial delay aggregation (v in TCN layout) -> token layout
__global__ void k_gather_s(const float* __restrict__ vT, const int* dly, const float* wts,
                           float* out, int Gd, int Nn, int L, int Kt, int gs, int ns)
{
  long long i = (long long)blockIdx.x * 256 + threadIdx.x;
  if (i >= (long long)Gd * Nn * L * DM) return;
  int d = (int)(i % DM); long long r = i / DM;
  int l = (int)(r % L); r /= L;
  int n = (int)(r % Nn); int g = (int)(r / Nn);
  int h = d / HD, c = d % HD;
  float s = 0.f;
  for (int kk = 0; kk < Kt; ++kk) {
    int nn = n + dly[g * 4 + kk]; if (nn >= Nn) nn -= Nn;
    s += wts[g * 4 + kk] * vT[((((long long)g * NH + h) * Nn + nn) * HD + c) * L + l];
  }
  int node = g * gs + n * ns;
  out[((long long)node * L + l) * DM + d] = s;
}

__global__ void k_ln(const float* __restrict__ x, const float* g, const float* b,
                     float* out, long long rows)
{
  long long r = (long long)blockIdx.x * 256 + threadIdx.x;
  if (r >= rows) return;
  const float* xp = x + r * DM;
  float m = 0.f;
  for (int d = 0; d < DM; ++d) m += xp[d];
  m /= DM;
  float v = 0.f;
  for (int d = 0; d < DM; ++d) { float dd = xp[d] - m; v += dd * dd; }
  v /= DM;
  float inv = rsqrtf(v + 1e-5f);
  for (int d = 0; d < DM; ++d) out[r * DM + d] = (xp[d] - m) * inv * g[d] + b[d];
}

__global__ void k_subtm(float* x, int L)
{
  long long i = (long long)blockIdx.x * 256 + threadIdx.x;
  if (i >= (long long)NNODE * DM) return;
  int d = (int)(i % DM); int n = (int)(i / DM);
  float m = 0.f;
  for (int l = 0; l < L; ++l) m += x[((long long)n * L + l) * DM + d];
  m /= L;
  for (int l = 0; l < L; ++l) x[((long long)n * L + l) * DM + d] -= m;
}

// trend_acc += conv1d_circ(trend_sum, Wtrend[5,128,3])
__global__ void k_trendconv(const float* __restrict__ ts, const float* __restrict__ Wt,
                            float* tacc5)
{
  long long i = (long long)blockIdx.x * 256 + threadIdx.x;
  if (i >= (long long)NNODE * LDEC * CINC) return;
  int c = (int)(i % CINC); long long r = i / CINC;
  int t = (int)(r % LDEC); int n = (int)(r / LDEC);
  float s = 0.f;
  for (int j = 0; j < 3; ++j) {
    int tt = (t - 1 + j + LDEC) % LDEC;
    const float* xp = ts + ((long long)n * LDEC + tt) * DM;
    for (int d = 0; d < DM; ++d) s += xp[d] * Wt[(c * DM + d) * 3 + j];
  }
  tacc5[i] += s;
}

// final projection + trend + inverse RevIN + slice -> d_out [350,24,5]
__global__ void k_final(const float* __restrict__ xd, const float* __restrict__ Wp,
                        const float* __restrict__ bp, const float* __restrict__ tacc5,
                        const float* aw, const float* ab,
                        const float* cmean, const float* cstd, float* out)
{
  long long i = (long long)blockIdx.x * 256 + threadIdx.x;
  if (i >= (long long)NNODE * LPRD * CINC) return;
  int c = (int)(i % CINC); long long r = i / CINC;
  int tp = (int)(r % LPRD); int n = (int)(r / LPRD);
  int t = tp + (LDEC - LPRD);
  float s = bp[c];
  const float* xp = xd + ((long long)n * LDEC + t) * DM;
  for (int d = 0; d < DM; ++d) s += xp[d] * Wp[d * CINC + c];
  float val = tacc5[((long long)n * LDEC + t) * CINC + c] + s;
  int col = n * CINC + c;
  val = (val - ab[c]) / (aw[c] + 1e-10f);
  val = val * cstd[col] + cmean[col];
  out[((long long)n * LPRD + tp) * CINC + c] = val;
}

// =====================================================================
// host orchestration
// =====================================================================
static inline dim3 g1(long long n) { return dim3((unsigned)((n + 255) / 256)); }

static void gemm(hipStream_t st, const float* A, const float* B, const float* bias,
                 const float* res, float* Y,
                 int M, int N, int K, int lda, int ldb, int ldy,
                 int transB, int bStride,
                 long long aB, long long yB, long long rB, int batch, int acc, int act)
{
  dim3 gr((N + 63) / 64, (M + 31) / 32, batch), bl(128);
  if (!transB) {
    if (res) k_gemm<0,0,1,1,1,0><<<gr, bl, 0, st>>>(A, B, bias, res, Y, M, N, K, lda, ldb, ldy, aB, yB, rB);
    else     k_gemm<0,0,1,1,0,0><<<gr, bl, 0, st>>>(A, B, bias, res, Y, M, N, K, lda, ldb, ldy, aB, yB, rB);
  } else if (bStride == 3) {
    if (acc) k_gemm<0,1,3,0,0,1><<<gr, bl, 0, st>>>(A, B, bias, res, Y, M, N, K, lda, ldb, ldy, aB, yB, rB);
    else     k_gemm<0,1,3,1,0,0><<<gr, bl, 0, st>>>(A, B, bias, res, Y, M, N, K, lda, ldb, ldy, aB, yB, rB);
  } else if (act == 2) {
    k_gemm<2,1,1,0,0,0><<<gr, bl, 0, st>>>(A, B, bias, res, Y, M, N, K, lda, ldb, ldy, aB, yB, rB);
  } else {
    k_gemm<0,1,1,0,1,0><<<gr, bl, 0, st>>>(A, B, bias, res, Y, M, N, K, lda, ldb, ldy, aB, yB, rB);
  }
}

struct MCtx {
  hipStream_t st;
  const float* const* prm;
  float *Qb, *Kb, *Vb, *QS, *KS, *VT, *Y0, *Y1;
  float *MC, *WTS; int* DLY;
};

// attn base layout (jax tree flatten, sorted keys): +0 Wk, +1 Wo, +2 Wq, +3 Wv,
// +4 bk, +5 bo, +6 bq, +7 bv ; tcn base: +0 W1, +1 W2, +2 b1, +3 b2
static void spatial_pass(const MCtx& c, const float* yin, float* yout,
                         int L, int Gd, int Nn, int gs, int ns, int ab, int tb)
{
  hipStream_t st = c.st;
  const int Mt = NNODE * L;
  gemm(st, yin, c.prm[ab + 2], c.prm[ab + 6], 0, c.Qb, Mt, DM, DM, DM, DM, DM, 0, 1, 0, 0, 0, 1, 0, 0);
  gemm(st, yin, c.prm[ab + 0], c.prm[ab + 4], 0, c.Kb, Mt, DM, DM, DM, DM, DM, 0, 1, 0, 0, 0, 1, 0, 0);
  gemm(st, yin, c.prm[ab + 3], c.prm[ab + 7], 0, c.Vb, Mt, DM, DM, DM, DM, DM, 0, 1, 0, 0, 0, 1, 0, 0);

  long long tS = (long long)Gd * DM * L * Nn;
  k_split_s<<<g1(tS), 256, 0, st>>>(c.Qb, c.QS, Gd, Nn, L, gs, ns);
  k_split_s<<<g1(tS), 256, 0, st>>>(c.Kb, c.KS, Gd, Nn, L, gs, ns);
  long long tV = (long long)Gd * NH * Nn * HD * L;
  k_split_vT<<<g1(tV), 256, 0, st>>>(c.Vb, c.VT, Gd, Nn, L, gs, ns);

  const int Ct = HD * L, batch = Gd * NH;
  const long long bS = (long long)Nn * Ct;
  // TCN: conv1 (VT->Qb), relu, conv2 (Qb->Kb), relu, residual relu (->Vb)
  for (int j = 2; j >= 0; --j) {
    int dd = 2 - j;
    gemm(st, c.VT, c.prm[tb + 0] + j, (j == 2) ? c.prm[tb + 2] : 0, 0,
         c.Qb + (long long)dd * Ct, Nn - dd, Ct, Ct, Ct, Ct, Ct, 1, 3, bS, bS, 0, batch, (j == 2) ? 0 : 1, 0);
  }
  long long tT = (long long)batch * Nn * Ct;
  k_relu<<<g1(tT), 256, 0, st>>>(c.Qb, tT);
  for (int j = 2; j >= 0; --j) {
    int dd = 2 - j;
    gemm(st, c.Qb, c.prm[tb + 1] + j, (j == 2) ? c.prm[tb + 3] : 0, 0,
         c.Kb + (long long)dd * Ct, Nn - dd, Ct, Ct, Ct, Ct, Ct, 1, 3, bS, bS, 0, batch, (j == 2) ? 0 : 1, 0);
  }
  k_relu<<<g1(tT), 256, 0, st>>>(c.Kb, tT);
  k_addrelu<<<g1(tT), 256, 0, st>>>(c.VT, c.Kb, c.Vb, tT);

  k_corr<<<Gd, 128, 0, st>>>(c.QS, c.KS, c.MC, DM * L, Nn);
  int Kt = (int)logf((float)Nn); if (Kt < 1) Kt = 1; if (Kt > 4) Kt = 4;
  k_topk<<<Gd, 32, 0, st>>>(c.MC, Nn, Kt, c.DLY, c.WTS);
  long long tG = (long long)Gd * Nn * L * DM;
  k_gather_s<<<g1(tG), 256, 0, st>>>(c.Vb, c.DLY, c.WTS, c.KS, Gd, Nn, L, Kt, gs, ns);
  gemm(st, c.KS, c.prm[ab + 1], c.prm[ab + 5], yin, yout, Mt, DM, DM, DM, DM, DM, 0, 1, 0, 0, 0, 1, 0, 0);
}

static void multi_corr(const MCtx& c, const float* xq, const float* xkv,
                       int Lq, int Lkv, int aa, int ac, int tb, float* out)
{
  hipStream_t st = c.st;
  const int Mq = NNODE * Lq, Mk = NNODE * Lkv;
  gemm(st, xq,  c.prm[aa + 2], c.prm[aa + 6], 0, c.Qb, Mq, DM, DM, DM, DM, DM, 0, 1, 0, 0, 0, 1, 0, 0);
  gemm(st, xkv, c.prm[aa + 0], c.prm[aa + 4], 0, c.Kb, Mk, DM, DM, DM, DM, DM, 0, 1, 0, 0, 0, 1, 0, 0);
  gemm(st, xkv, c.prm[aa + 3], c.prm[aa + 7], 0, c.Vb, Mk, DM, DM, DM, DM, DM, 0, 1, 0, 0, 0, 1, 0, 0);

  long long tS = (long long)NNODE * DM * Lq;
  k_split_t<<<g1(tS), 256, 0, st>>>(c.Qb, c.QS, NNODE, Lq, Lq);
  k_split_t<<<g1(tS), 256, 0, st>>>(c.Kb, c.KS, NNODE, Lkv, Lq);  // zero-pad if Lkv<Lq
  k_split_t<<<g1(tS), 256, 0, st>>>(c.Vb, c.VT, NNODE, Lkv, Lq);

  k_corr<<<NNODE, 128, 0, st>>>(c.QS, c.KS, c.MC, DM, Lq);
  int Kt = (int)logf((float)Lq); if (Kt < 1) Kt = 1; if (Kt > 4) Kt = 4;
  k_topk<<<NNODE, 32, 0, st>>>(c.MC, Lq, Kt, c.DLY, c.WTS);
  k_gather_t<<<g1(tS), 256, 0, st>>>(c.VT, c.DLY, c.WTS, c.KS, NNODE, Lq, Kt);
  gemm(st, c.KS, c.prm[aa + 1], c.prm[aa + 5], 0, c.Y0, Mq, DM, DM, DM, DM, DM, 0, 1, 0, 0, 0, 1, 0, 0);

  spatial_pass(c, c.Y0, c.Y1, Lq, 7, 50, 50, 1, ac, tb);    // within-group
  spatial_pass(c, c.Y1, out,  Lq, 50, 7, 1, 50, ac, tb);    // across-group
}

extern "C" void kernel_launch(void* const* d_in, const int* in_sizes, int n_in,
                              void* d_out, int out_size, void* d_ws, size_t ws_size,
                              hipStream_t stream)
{
  (void)in_sizes; (void)out_size; (void)ws_size;
  hipStream_t st = stream;

  // param leaves in jax tree_flatten (sorted-key) order, after the 4 data inputs
  const float* prm[112];
  for (int i = 0; i < n_in && i < 112; ++i) prm[i] = (const float*)d_in[i];

  float* W = (float*)d_ws;
  size_t off = 0;
  auto alloc = [&](size_t nfl) { size_t r = off; off += (nfl + 63) & ~(size_t)63; return r; };
  const size_t BIGF = (size_t)NNODE * LDEC * DM;   // 3,225,600 floats

  float* xA  = W + alloc(BIGF);
  float* xB  = W + alloc(BIGF);
  float* ENC = W + alloc(BIGF);
  float* Qb  = W + alloc(BIGF);   // Qb..QS contiguous: FFN hidden aliases this range
  float* Kb  = W + alloc(BIGF);
  float* Vb  = W + alloc(BIGF);
  float* QS  = W + alloc(BIGF);
  float* KS  = W + alloc(BIGF);
  float* VT  = W + alloc(BIGF);
  float* Y0  = W + alloc(BIGF);
  float* Y1  = W + alloc(BIGF);
  float* Y2  = W + alloc(BIGF);
  float* TS  = W + alloc(BIGF);
  float* HID = Qb;                // 4*BIGF >= 25200*512

  float* xeN   = W + alloc((size_t)NNODE * SEQL * CINC);
  float* seasN = W + alloc((size_t)NNODE * SEQL * CINC);
  float* trndN = W + alloc((size_t)NNODE * SEQL * CINC);
  float* sinit = W + alloc((size_t)NNODE * LDEC * CINC);
  float* tacc5 = W + alloc((size_t)NNODE * LDEC * CINC);
  float* cmean = W + alloc(NNODE * CINC);
  float* cstd  = W + alloc(NNODE * CINC);
  float* cmn2  = W + alloc(NNODE * CINC);
  float* markD = W + alloc(LDEC * MRK);
  float* MCb   = W + alloc((size_t)NNODE * LDEC);
  float* WTS   = W + alloc(NNODE * 4);
  int*   DLY   = (int*)(W + alloc(NNODE * 4));

  MCtx c; c.st = st; c.prm = prm;
  c.Qb = Qb; c.Kb = Kb; c.Vb = Vb; c.QS = QS; c.KS = KS; c.VT = VT;
  c.Y0 = Y0; c.Y1 = Y1; c.MC = MCb; c.WTS = WTS; c.DLY = DLY;

  // ---- preprocessing: RevIN + decomposition + inits ----
  k_prep<<<g1(NNODE * CINC), 256, 0, st>>>(prm[0], prm[5], prm[4], xeN, cmean, cstd, cmn2);
  k_decomp<<<g1(NNODE * CINC), 256, 0, st>>>(xeN, seasN, trndN, 0, NNODE, SEQL, CINC);
  k_seasinit<<<g1((long long)NNODE * LDEC * CINC), 256, 0, st>>>(seasN, trndN, cmn2, sinit, tacc5);

  // ---- encoder ----
  k_embed<<<g1((long long)NNODE * SEQL * DM), 256, 0, st>>>(xeN, prm[55], prm[1], prm[54], prm[56], xA, SEQL);
  for (int l = 0; l < 2; ++l) {
    int fW1, fW2, aa, ac, tb;
    if (l == 0) { fW1 = 57; fW2 = 58; aa = 59; ac = 67; tb = 75; }
    else        { fW1 = 79; fW2 = 80; aa = 81; ac = 89; tb = 97; }
    multi_corr(c, xA, xA, SEQL, SEQL, aa, ac, tb, Y2);
    long long tE = (long long)NNODE * SEQL * DM;
    k_add<<<g1(tE), 256, 0, st>>>(xA, Y2, xB, tE);
    k_decomp<<<g1(NNODE * DM), 256, 0, st>>>(xB, xA, 0, 0, NNODE, SEQL, DM);
    const int Me = NNODE * SEQL;
    gemm(st, xA, prm[fW1], 0, 0, HID, Me, DFF, DM, DM, DM, DFF, 1, 1, 0, 0, 0, 1, 0, 2);  // gelu
    gemm(st, HID, prm[fW2], 0, xA, xB, Me, DM, DFF, DFF, DFF, DM, 1, 1, 0, 0, 0, 1, 0, 0);
    k_decomp<<<g1(NNODE * DM), 256, 0, st>>>(xB, xA, 0, 0, NNODE, SEQL, DM);
  }
  k_ln<<<g1((long long)NNODE * SEQL), 256, 0, st>>>(xA, prm[102], prm[101], ENC, (long long)NNODE * SEQL);
  k_subtm<<<g1(NNODE * DM), 256, 0, st>>>(ENC, SEQL);

  // ---- decoder ----
  k_markd<<<g1(LDEC * MRK), 256, 0, st>>>(prm[1], prm[3], markD);
  k_embed<<<g1((long long)NNODE * LDEC * DM), 256, 0, st>>>(sinit, prm[7], markD, prm[6], prm[8], xA, LDEC);
  k_fill<<<g1((long long)BIGF), 256, 0, st>>>(TS, (long long)BIGF, 0.f);

  long long tD = (long long)NNODE * LDEC * DM;
  // self multi-corr
  multi_corr(c, xA, xA, LDEC, LDEC, 32, 40, 48, Y2);
  k_add<<<g1(tD), 256, 0, st>>>(xA, Y2, xB, tD);
  k_decomp<<<g1(NNODE * DM), 256, 0, st>>>(xB, xA, 0, TS, NNODE, LDEC, DM);
  // cross multi-corr against encoder output (k/v zero-padded 48->72)
  multi_corr(c, xA, ENC, LDEC, SEQL, 10, 18, 26, Y2);
  k_add<<<g1(tD), 256, 0, st>>>(xA, Y2, xB, tD);
  k_decomp<<<g1(NNODE * DM), 256, 0, st>>>(xB, xA, 0, TS, NNODE, LDEC, DM);
  // FFN
  const int Md = NNODE * LDEC;
  gemm(st, xA, prm[30], 0, 0, HID, Md, DFF, DM, DM, DM, DFF, 1, 1, 0, 0, 0, 1, 0, 2);
  gemm(st, HID, prm[31], 0, xA, xB, Md, DM, DFF, DFF, DFF, DM, 1, 1, 0, 0, 0, 1, 0, 0);
  k_decomp<<<g1(NNODE * DM), 256, 0, st>>>(xB, xA, 0, TS, NNODE, LDEC, DM);
  // trend projection conv
  k_trendconv<<<g1((long long)NNODE * LDEC * CINC), 256, 0, st>>>(TS, prm[9], tacc5);

  // ---- output head ----
  k_ln<<<g1((long long)Md), 256, 0, st>>>(xA, prm[53], prm[52], xB, (long long)Md);
  k_subtm<<<g1(NNODE * DM), 256, 0, st>>>(xB, LDEC);
  k_final<<<g1((long long)NNODE * LPRD * CINC), 256, 0, st>>>(xB, prm[103], prm[104], tacc5,
                                                             prm[5], prm[4], cmean, cstd,
                                                             (float*)d_out);
}
